// Conv2d_3049426780812
// MI455X (gfx1250) — matrix operand order repaired and checked by reading, not compiled, not run on hardware
//
#include <hip/hip_runtime.h>
#include <cstdint>

// ---------------------------------------------------------------------------
// 3x3 conv (N=32, C=128, K=256, 56x56, pad=1, stride=1), fp32 in/out.
// Implicit GEMM, split-bf16 ("3xBF16") on v_wmma_f32_16x16x32_bf16,
// Tensor Data Mover (tensor_load_to_lds) + double-buffered LDS pipeline.
// ---------------------------------------------------------------------------

typedef __attribute__((ext_vector_type(16))) __bf16 v16bf;
typedef __attribute__((ext_vector_type(8)))  float  v8f;

#define N_  32
#define C_  128
#define K_  256
#define H_  56
#define W_  56
#define HP  58   // padded height  (1 + 56 + 1)
#define WP  66   // padded width   (>= 1 + 63 + 2 for the x-overscan tile)

#define NSTEP 36 // 9 taps x 4 chunks of 32 channels

__device__ __forceinline__ uint16_t bf16_rne(float x) {
  uint32_t u = __float_as_uint(x);
  u += 0x7FFFu + ((u >> 16) & 1u);       // round-to-nearest-even to bf16
  return (uint16_t)(u >> 16);
}

__device__ __forceinline__ void split_bf16(float x, uint16_t& hi, uint16_t& lo) {
  hi = bf16_rne(x);
  float fhi = __uint_as_float((uint32_t)hi << 16);
  lo = bf16_rne(x - fhi);
}

// ---- weight prep: OIHW fp32 -> [tap][ko][c] bf16 hi/lo ---------------------
__global__ void prep_w(const float* __restrict__ w,
                       uint16_t* __restrict__ whi, uint16_t* __restrict__ wlo) {
  int tid = blockIdx.x * blockDim.x + threadIdx.x;
  if (tid >= 9 * K_ * C_) return;
  int c  = tid % C_;
  int ko = (tid / C_) % K_;
  int t  = tid / (K_ * C_);
  int r = t / 3, s = t % 3;
  float v = w[((ko * C_ + c) * 3 + r) * 3 + s];
  uint16_t h, l; split_bf16(v, h, l);
  whi[tid] = h; wlo[tid] = l;
}

// ---- input prep: NCHW fp32 -> zero-padded NHWC bf16 hi/lo [n][HP][WP][C] ---
__global__ void prep_in(const float* __restrict__ in,
                        uint16_t* __restrict__ ihi, uint16_t* __restrict__ ilo) {
  int tid = blockIdx.x * blockDim.x + threadIdx.x;
  if (tid >= N_ * HP * WP * C_) return;
  int c   = tid % C_;
  int rem = tid / C_;
  int xp  = rem % WP; rem /= WP;
  int yp  = rem % HP;
  int n   = rem / HP;
  float v = 0.0f;
  if (yp >= 1 && yp <= H_ && xp >= 1 && xp <= W_) {
    v = in[((n * C_ + c) * H_ + (yp - 1)) * W_ + (xp - 1)];
  }
  uint16_t h, l; split_bf16(v, h, l);
  ihi[tid] = h; ilo[tid] = l;
}

// ---------------------------------------------------------------------------
// Tensor Data Mover descriptor build + issue (clang-23 6-arg builtin form:
//   (u32x4 g0, i32x8 g1, i32x4 g2, i32x4 g3, i32x8 g4, i32 cpol) ).
// D# packing per CDNA5 ISA.
// ---------------------------------------------------------------------------
#if __has_builtin(__builtin_amdgcn_tensor_load_to_lds)
#define HAVE_TDM 1
typedef unsigned int u32x4 __attribute__((ext_vector_type(4)));
typedef int          i32x4 __attribute__((ext_vector_type(4)));
typedef int          i32x8 __attribute__((ext_vector_type(8)));

__device__ __forceinline__ void tdm_load(const uint16_t* gptr, const uint16_t* lds_ptr,
                                         uint32_t tile_d0, uint32_t tile_d1,
                                         uint32_t tile_d2,
                                         uint64_t stride0, uint64_t stride1) {
  const uint64_t ga  = (uint64_t)(uintptr_t)gptr;
  const uint32_t lds = (uint32_t)(uintptr_t)lds_ptr;  // LDS aperture: low 32 = offset
  const uint32_t td0 = 0x7FFFFFFFu, td1 = 0x7FFFFFFFu; // in-bounds by construction

  u32x4 g0;
  g0[0] = 1u;                                          // count=1, user descriptor
  g0[1] = lds;                                         // lds_addr [63:32]
  g0[2] = (uint32_t)ga;                                // global_addr [95:64]
  g0[3] = (uint32_t)((ga >> 32) & 0x01FFFFFFu) | 0x80000000u; // ga[56:32] | type=2

  i32x8 g1;
  g1[0] = (int)(1u << 16);                             // data_size=1 (2 bytes)
  g1[1] = (int)((td0 & 0xFFFFu) << 16);                // tensor_dim0[15:0]
  g1[2] = (int)((td0 >> 16) | ((td1 & 0xFFFFu) << 16));// dim0[31:16] | dim1[15:0]
  g1[3] = (int)((td1 >> 16) | (tile_d0 << 16));        // dim1[31:16] | tile_dim0
  g1[4] = (int)((tile_d1 & 0xFFFFu) | (tile_d2 << 16));// tile_dim1 | tile_dim2
  g1[5] = (int)(uint32_t)(stride0 & 0xFFFFFFFFu);      // tensor_dim0_stride[31:0]
  g1[6] = (int)(uint32_t)(((stride0 >> 32) & 0xFFFFu) |
                          ((stride1 & 0xFFFFu) << 16));// str0[47:32] | str1[15:0]
  g1[7] = (int)(uint32_t)((stride1 >> 16) & 0xFFFFFFFFu); // str1[47:16]

  i32x4 g2; g2[0] = 0x7FFFFFFF; g2[1] = 0; g2[2] = 0; g2[3] = 0; // tensor_dim2 big
  i32x4 g3; g3[0] = 0; g3[1] = 0; g3[2] = 0; g3[3] = 0;
  i32x8 g4; g4[0] = 0; g4[1] = 0; g4[2] = 0; g4[3] = 0;
            g4[4] = 0; g4[5] = 0; g4[6] = 0; g4[7] = 0;

  __builtin_amdgcn_tensor_load_to_lds(g0, g1, g2, g3, g4, 0);
}

__device__ __forceinline__ void tdm_wait() {
#if __has_builtin(__builtin_amdgcn_s_wait_tensorcnt)
  __builtin_amdgcn_s_wait_tensorcnt(0);
#else
  asm volatile("s_wait_tensorcnt 0x0" ::: "memory");
#endif
}
#else
#define HAVE_TDM 0
#endif

// ---- main kernel: 128-chan x 128-pixel (16x by 8y) tile per block ----------
// 8 waves in 2(M) x 4(N): each wave owns 64 chans x 32 pixels = 4x2 WMMA tiles.
// Pipeline: TDM fills LDS buffer (k+1) while all waves run WMMA on buffer k.
__global__ __launch_bounds__(256, 2)
void conv_wmma(const uint16_t* __restrict__ whi, const uint16_t* __restrict__ wlo,
               const uint16_t* __restrict__ ihi, const uint16_t* __restrict__ ilo,
               float* __restrict__ out) {
  // [buffer][hi/lo][128 rows x 32 bf16] -> 64 KB total of 320 KB WGP LDS
  __shared__ __align__(16) uint16_t sA[2][2][128 * 32];
  __shared__ __align__(16) uint16_t sB[2][2][128 * 32];

  const int tid = threadIdx.x;
  int bx = blockIdx.x;
  const int kb = bx & 1;  bx >>= 1;            // channel block (2 x 128)
  const int xb = bx & 3;  bx >>= 2;            // x block (4 x 16, overscan)
  const int yb = bx % 7;                       // y block (7 x 8)
  const int n  = bx / 7;                       // batch
  const int x0 = xb * 16, y0 = yb * 8;

  const int lane = tid & 31;
  const int wave = tid >> 5;
  const int wm = wave >> 2;                    // 0..1  (64 channels each)
  const int wn = wave & 3;                     // 0..3  (32 pixels each)

  v8f acc[4][2] = {};                          // fp32 accumulators

  // ---- tile loader: step s in [0,36) -> buffer nb ----
  auto load_tiles = [&](int s, int nb) {
    const int t  = s >> 2, cb = s & 3;
    const int dy = t / 3 - 1, dx = t % 3 - 1;
#if HAVE_TDM
    if (wave == 0) {                           // one wave drives the TDM
      const int aoff = (t * K_ + kb * 128) * C_ + cb * 32;
      const int boff = ((n * HP + (y0 + dy + 1)) * WP + (x0 + dx + 1)) * C_ + cb * 32;
      // A: 2D tile 32(c) x 128(ko), row stride C_
      tdm_load(whi + aoff, sA[nb][0], 32, 128, 0, C_, 0);
      tdm_load(wlo + aoff, sA[nb][1], 32, 128, 0, C_, 0);
      // B: 3D tile 32(c) x 16(x) x 8(y), strides C_ and WP*C_
      tdm_load(ihi + boff, sB[nb][0], 32, 16, 8, C_, (uint64_t)WP * C_);
      tdm_load(ilo + boff, sB[nb][1], 32, 16, 8, C_, (uint64_t)WP * C_);
    }
#else
    // fallback: per-thread 16B segments global->VGPR->LDS
    float4* a4h = reinterpret_cast<float4*>(sA[nb][0]);
    float4* a4l = reinterpret_cast<float4*>(sA[nb][1]);
    float4* b4h = reinterpret_cast<float4*>(sB[nb][0]);
    float4* b4l = reinterpret_cast<float4*>(sB[nb][1]);
    #pragma unroll
    for (int it = 0; it < 2; ++it) {
      const int sidx = tid + it * 256;
      const int row  = sidx >> 2;
      const int seg  = sidx & 3;
      const int wa = (t * K_ + kb * 128 + row) * C_ + cb * 32 + seg * 8;
      a4h[row * 4 + seg] = *reinterpret_cast<const float4*>(whi + wa);
      a4l[row * 4 + seg] = *reinterpret_cast<const float4*>(wlo + wa);
      const int yy = row >> 4, xx = row & 15;
      const int ia = ((n * HP + (y0 + yy + dy + 1)) * WP + (x0 + xx + dx + 1)) * C_
                     + cb * 32 + seg * 8;
      b4h[row * 4 + seg] = *reinterpret_cast<const float4*>(ihi + ia);
      b4l[row * 4 + seg] = *reinterpret_cast<const float4*>(ilo + ia);
    }
#endif
  };

  auto finish_loads = [&]() {
#if HAVE_TDM
    if (wave == 0) tdm_wait();
#endif
  };

  // ---- prologue: fill buffer 0 ----
  load_tiles(0, 0);
  finish_loads();
  __syncthreads();

  for (int s = 0; s < NSTEP; ++s) {
    const int cur = s & 1, nxt = cur ^ 1;
    if (s + 1 < NSTEP) load_tiles(s + 1, nxt);   // overlap with compute

    // ---- fragments (ISA 16-bit A 16x32 / B 32x16 layouts) ----
    const float4* a4h = reinterpret_cast<const float4*>(sA[cur][0]);
    const float4* a4l = reinterpret_cast<const float4*>(sA[cur][1]);
    const float4* b4h = reinterpret_cast<const float4*>(sB[cur][0]);
    const float4* b4l = reinterpret_cast<const float4*>(sB[cur][1]);
    union Frag { v16bf v; float4 f[2]; };
    Frag ah[4], al[4], bh[2], bl[2];
    #pragma unroll
    for (int mt = 0; mt < 4; ++mt) {
      const int row = wm * 64 + mt * 16 + (lane & 15);
      const int i0  = row * 4 + (lane >> 4);       // K seg 0/8, then +16
      ah[mt].f[0] = a4h[i0]; ah[mt].f[1] = a4h[i0 + 2];
      al[mt].f[0] = a4l[i0]; al[mt].f[1] = a4l[i0 + 2];
    }
    #pragma unroll
    for (int nt = 0; nt < 2; ++nt) {
      const int col = wn * 32 + nt * 16 + (lane & 15);
      const int i0  = col * 4 + (lane >> 4);       // K seg 0/8, then +16
      bh[nt].f[0] = b4h[i0]; bh[nt].f[1] = b4h[i0 + 2];
      bl[nt].f[0] = b4l[i0]; bl[nt].f[1] = b4l[i0 + 2];
    }

    // ---- 3xBF16 WMMA: hi*hi + hi*lo + lo*hi ----
    #pragma unroll
    for (int mt = 0; mt < 4; ++mt) {
      #pragma unroll
      for (int nt = 0; nt < 2; ++nt) {
        acc[mt][nt] = __builtin_amdgcn_wmma_f32_16x16x32_bf16(
            false, ah[mt].v, false, bh[nt].v, (short)0, acc[mt][nt], false, false);
        acc[mt][nt] = __builtin_amdgcn_wmma_f32_16x16x32_bf16(
            false, ah[mt].v, false, bl[nt].v, (short)0, acc[mt][nt], false, false);
        acc[mt][nt] = __builtin_amdgcn_wmma_f32_16x16x32_bf16(
            false, al[mt].v, false, bh[nt].v, (short)0, acc[mt][nt], false, false);
      }
    }

    finish_loads();                              // TDM for (s+1) complete
    __syncthreads();                             // everyone done reading 'cur'
  }

  // ---- store: C/D layout = lane%16 -> N(pixel), vgpr j + (lane/16)*8 -> M ----
  #pragma unroll
  for (int mt = 0; mt < 4; ++mt) {
    const int m = kb * 128 + wm * 64 + mt * 16 + (lane >> 4) * 8;
    #pragma unroll
    for (int nt = 0; nt < 2; ++nt) {
      const int px = wn * 32 + nt * 16 + (lane & 15);
      const int y = y0 + (px >> 4);
      const int x = x0 + (px & 15);
      if (x < W_) {
        #pragma unroll
        for (int j = 0; j < 8; ++j) {
          out[(((long)n * K_ + (m + j)) * H_ + y) * W_ + x] = acc[mt][nt][j];
        }
      }
    }
  }
}

// ---------------------------------------------------------------------------
extern "C" void kernel_launch(void* const* d_in, const int* in_sizes, int n_in,
                              void* d_out, int out_size, void* d_ws, size_t ws_size,
                              hipStream_t stream) {
  (void)in_sizes; (void)n_in; (void)out_size; (void)ws_size;
  const float* in = (const float*)d_in[0];
  const float* w  = (const float*)d_in[1];
  float* out = (float*)d_out;

  // workspace layout (uint16 elements): whi | wlo | ihi | ilo  (~61 MB total)
  const size_t WN = (size_t)9 * K_ * C_;        //   294,912
  const size_t IN = (size_t)N_ * HP * WP * C_;  // ~15.68 M
  uint16_t* whi = (uint16_t*)d_ws;
  uint16_t* wlo = whi + WN;
  uint16_t* ihi = wlo + WN;
  uint16_t* ilo = ihi + IN;

  hipLaunchKernelGGL(prep_w,  dim3((unsigned)((WN + 255) / 256)), dim3(256), 0, stream,
                     w, whi, wlo);
  hipLaunchKernelGGL(prep_in, dim3((unsigned)((IN + 255) / 256)), dim3(256), 0, stream,
                     in, ihi, ilo);

  // grid: 2 chan-blocks x 4 x-blocks x 7 y-blocks x 32 batch = 1792 blocks
  hipLaunchKernelGGL(conv_wmma, dim3(2 * 4 * 7 * N_), dim3(256), 0, stream,
                     whi, wlo, ihi, ilo, out);
}
